// MultiHeadAttentionBlock_22290880266251
// MI455X (gfx1250) — matrix operand order repaired and checked by reading, not compiled, not run on hardware
//
#include <hip/hip_runtime.h>
#include <hip/hip_bf16.h>

// MultiHeadAttentionBlock on MI455X (gfx1250), bf16 WMMA everywhere.
// B=2, S=2048, D_MODEL=1024, H=16, D_K=64.
// Pipeline: convert f32->bf16 (x7) -> 3 projection GEMMs (TDM-staged W panel)
//           -> flash attention -> output projection GEMM (f32 out).

#define S_LEN 2048
#define DM    1024
#define NH    16
#define DK    64

typedef __bf16 bf16;
typedef __attribute__((ext_vector_type(16))) __bf16 v16bf;
typedef __attribute__((ext_vector_type(8)))  __bf16 v8bf;
typedef __attribute__((ext_vector_type(8)))  float  v8f;
typedef __attribute__((ext_vector_type(8)))  float  f32x8;
typedef __attribute__((ext_vector_type(4)))  unsigned u32x4;
typedef __attribute__((ext_vector_type(4)))  unsigned tdm_g0_t;
typedef __attribute__((ext_vector_type(8)))  int      tdm_g1_t;
typedef __attribute__((ext_vector_type(4)))  int      tdm_g2_t;

#if defined(__gfx1250__) && __has_builtin(__builtin_amdgcn_tensor_load_to_lds) && \
    __has_builtin(__builtin_amdgcn_s_wait_tensorcnt)
#define HAVE_TDM 1
#else
#define HAVE_TDM 0
#endif

union BF16x16 { v16bf v; u32x4 q[2]; };
union BF16x8  { v8bf  v; u32x4 q;    };

static __device__ __forceinline__ bf16 f2bf(float f) {
  // round-to-nearest-even f32 -> bf16 without relying on __bf16 conversions
  unsigned u = __float_as_uint(f);
  u += 0x7fffu + ((u >> 16) & 1u);
  unsigned short h = (unsigned short)(u >> 16);
  bf16 b;
  __builtin_memcpy(&b, &h, 2);
  return b;
}

static __device__ __forceinline__ v8f wmma_bf16(const BF16x16& a, const BF16x16& b, v8f c) {
  // D = A(16x32 bf16) * B(32x16 bf16) + C(16x16 f32)
  return __builtin_amdgcn_wmma_f32_16x16x32_bf16(false, a.v, false, b.v, (short)0, c,
                                                 false, false);
}

#if HAVE_TDM
// Issue a TDM load of a [tile_rows x tile_cols] bf16 tile (row stride = DM
// elements) from global memory into LDS at byte offset lds_byte.
// D# per CDNA5 ISA ch.8: group0 = {count, lds_addr, global_addr, type=2},
// group1 = {data_size=2B, tensor_dim0/1, tile_dim0/1, tensor_dim0_stride}.
static __device__ __forceinline__ void tdm_load_tile(unsigned lds_byte,
                                                     const bf16* gsrc,
                                                     int tile_cols, int tile_rows) {
  unsigned long long ga = (unsigned long long)(size_t)gsrc;
  tdm_g0_t g0;
  g0[0] = 1u;                                            // count=1, user mode
  g0[1] = lds_byte;                                      // lds_addr
  g0[2] = (unsigned)(ga & 0xffffffffull);                // global_addr[31:0]
  g0[3] = (unsigned)((ga >> 32) & 0x01ffffffull) | (2u << 30);  // addr[56:32] | type=2
  tdm_g1_t g1 = {0, 0, 0, 0, 0, 0, 0, 0};
  g1[0] = (int)(1u << 16);                               // data_size = 1 -> 2 bytes
  g1[1] = (int)((unsigned)(DM & 0xffff) << 16);          // tensor_dim0 = DM
  g1[2] = (int)(((unsigned)DM >> 16) | ((unsigned)(DM & 0xffff) << 16));  // dim0 hi | dim1 lo
  g1[3] = (int)(((unsigned)DM >> 16) | ((unsigned)tile_cols << 16));      // dim1 hi | tile_dim0
  g1[4] = tile_rows & 0xffff;                            // tile_dim1 (tile_dim2 = 0)
  g1[5] = DM;                                            // tensor_dim0_stride lo
  tdm_g2_t gz = {0, 0, 0, 0};
#if __clang_major__ >= 23
  tdm_g1_t gz8 = {0, 0, 0, 0, 0, 0, 0, 0};
  __builtin_amdgcn_tensor_load_to_lds(g0, g1, gz, gz, gz8, 0);
#else
  __builtin_amdgcn_tensor_load_to_lds(g0, g1, gz, gz, 0);
#endif
}
#endif

// ---------------------------------------------------------------------------
// Elementwise f32 -> bf16 conversion (bandwidth bound). One thread = 8 elems.
// ---------------------------------------------------------------------------
__global__ __launch_bounds__(256) void cvt_f32_bf16_kernel(const float* __restrict__ src,
                                                           bf16* __restrict__ dst,
                                                           int n8) {
  const int i = blockIdx.x * 256 + threadIdx.x;
  if (i < n8) {
    f32x8 x = ((const f32x8*)src)[i];
    BF16x8 p;
#pragma unroll
    for (int r = 0; r < 8; ++r) p.v[r] = f2bf(x[r]);
    ((u32x4*)dst)[i] = p.q;
  }
}

// ---------------------------------------------------------------------------
// GEMM (all-bf16 operands): Y[m,e] = sum_d X[m,d] * W[e,d]
// m = b*S + s (4096 rows), e (1024 cols). One wave computes a 32x64 tile
// (2 A-frags x 4 B-frags, 8 WMMA/k-step). All 8 waves of a block share one
// 64-row W panel: it is staged into LDS in 64x128 chunks, double-buffered via
// the Tensor Data Mover (TENSORcnt), with a cooperative-copy fallback.
// MODE 0: Q -> out[b][h][s][dk] bf16, scaled by 1/sqrt(DK)
// MODE 1: K -> out[b][h][s][dk] bf16
// MODE 2: V -> out[b][h][dk][s] bf16 (transposed for contiguous PV B-frags)
// MODE 3: output projection -> out[m][e] f32
// ---------------------------------------------------------------------------
template <int MODE>
__global__ __launch_bounds__(256) void mha_gemm_kernel(const bf16* __restrict__ X,
                                                       const bf16* __restrict__ W,
                                                       void* __restrict__ outp) {
  __shared__ bf16 lbuf[2 * 64 * 128];                  // 2 x 16 KB chunks

  const int lane = threadIdx.x & 31;
  const int wave = threadIdx.x >> 5;
  const int wid  = blockIdx.x * 8 + wave;              // 2048 waves total
  const int mt   = wid & 127;                          // 128 row tiles of 32
  const int nt   = wid >> 7;                           // 16 col groups of 64 (uniform per block)
  const int m0   = mt << 5;
  const int n0   = nt << 6;
  const int half = lane >> 4;
  const int ln   = lane & 15;

  const bf16* xrow0 = X + (size_t)(m0 + ln) * DM;
  const bf16* xrow1 = X + (size_t)(m0 + 16 + ln) * DM;

  v8f acc0[4] = {v8f{}, v8f{}, v8f{}, v8f{}};
  v8f acc1[4] = {v8f{}, v8f{}, v8f{}, v8f{}};

#if HAVE_TDM
  const unsigned lds_base = (unsigned)(size_t)(void*)lbuf;  // low 32 bits = LDS offset
  if (wave == 0) tdm_load_tile(lds_base, W + (size_t)n0 * DM, 128, 64);
#endif

  for (int c0 = 0; c0 < DM; c0 += 128) {
    const int buf = (c0 >> 7) & 1;
#if HAVE_TDM
    if (wave == 0) {
      if (c0 + 128 < DM) {
        tdm_load_tile(lds_base + (unsigned)((buf ^ 1) * 64 * 128 * 2),
                      W + (size_t)n0 * DM + c0 + 128, 128, 64);
        __builtin_amdgcn_s_wait_tensorcnt(1);   // chunk c0 done (TDM is in-order)
      } else {
        __builtin_amdgcn_s_wait_tensorcnt(0);
      }
    }
    __syncthreads();
    const bf16* lw = lbuf + buf * (64 * 128);
#else
    // fallback: cooperative global->LDS copy of this 64x128 chunk (single buffer)
    (void)buf;
    __syncthreads();
    {
      const int tid = (int)threadIdx.x;
      const int r  = tid >> 2;          // 64 rows
      const int qd = tid & 3;           // 4 x 32 bf16 per row
      const u32x4* s = (const u32x4*)(W + (size_t)(n0 + r) * DM + c0 + qd * 32);
      u32x4* d = (u32x4*)(lbuf + r * 128 + qd * 32);
#pragma unroll
      for (int i = 0; i < 4; ++i) d[i] = s[i];
    }
    __syncthreads();
    const bf16* lw = lbuf;
#endif

#pragma unroll
    for (int dd = 0; dd < 128; dd += 32) {
      const int d0 = c0 + dd;
      __builtin_prefetch(xrow0 + d0 + 128, 0, 3);
      __builtin_prefetch(xrow1 + d0 + 128, 0, 3);
      // A fragments: rows m0+ln / m0+16+ln, K(d)=d0+8h..+7 and d0+16+8h..+7
      BF16x16 a0, a1;
      a0.q[0] = *(const u32x4*)(xrow0 + d0 + 8 * half);
      a0.q[1] = *(const u32x4*)(xrow0 + d0 + 16 + 8 * half);
      a1.q[0] = *(const u32x4*)(xrow1 + d0 + 8 * half);
      a1.q[1] = *(const u32x4*)(xrow1 + d0 + 16 + 8 * half);
#pragma unroll
      for (int j = 0; j < 4; ++j) {
        // B fragment from LDS: row e-local = 16j+ln, K(d) = dd+8h..+7 and dd+16+8h..+7
        const bf16* lwj = lw + (j * 16 + ln) * 128 + dd + 8 * half;
        BF16x16 bf;
        bf.q[0] = *(const u32x4*)(lwj);
        bf.q[1] = *(const u32x4*)(lwj + 16);
        acc0[j] = wmma_bf16(a0, bf, acc0[j]);
        acc1[j] = wmma_bf16(a1, bf, acc1[j]);
      }
    }
    __syncthreads();   // everyone done reading before this buffer is overwritten
  }

  // C layout: lane holds col n0+16j+ln; VGPR r holds row base + r + 8*half.
  const int b = m0 >> 11;                              // 2048 rows per batch
#pragma unroll
  for (int t = 0; t < 2; ++t) {
    const v8f* acc = t ? acc1 : acc0;
    const int s0r  = (m0 & 2047) + t * 16 + 8 * half;
#pragma unroll
    for (int j = 0; j < 4; ++j) {
      const int e  = n0 + j * 16 + ln;
      const int hh = e >> 6;
      const int dk = e & 63;
      if (MODE == 3) {
        float* Y = (float*)outp;
#pragma unroll
        for (int r = 0; r < 8; ++r)
          Y[(size_t)(m0 + t * 16 + 8 * half + r) * DM + e] = acc[j][r];
      } else if (MODE == 2) {
        // transposed: 8 consecutive s per lane -> one 16B store
        bf16* out = (bf16*)outp;
        size_t base = (((size_t)b * NH + hh) * DK + dk) * (size_t)S_LEN + s0r;
        BF16x8 p;
#pragma unroll
        for (int r = 0; r < 8; ++r) p.v[r] = f2bf(acc[j][r]);
        *(u32x4*)(out + base) = p.q;
      } else {
        bf16* out = (bf16*)outp;
        const float sc = (MODE == 0) ? 0.125f : 1.0f;  // fold 1/sqrt(64) into Q
        size_t base = (((size_t)b * NH + hh) * S_LEN + s0r) * (size_t)DK + dk;
#pragma unroll
        for (int r = 0; r < 8; ++r) out[base + (size_t)r * DK] = f2bf(acc[j][r] * sc);
      }
    }
  }
}

// ---------------------------------------------------------------------------
// Flash attention (causal). One wave owns a 16-query tile of one head.
// Scores computed transposed: St = K_tile(16k x 64d) * Q^T(64d x 16q), so the
// C-layout of St IS the A-fragment layout for P in the P*V matmul.
// Online softmax across key chunks of 32.
// ---------------------------------------------------------------------------
__global__ __launch_bounds__(256) void mha_attn_kernel(const bf16* __restrict__ Q,
                                                       const bf16* __restrict__ K,
                                                       const bf16* __restrict__ Vt,
                                                       bf16* __restrict__ O) {
  const int lane = threadIdx.x & 31;
  const int wave = threadIdx.x >> 5;
  const int wid  = blockIdx.x * 8 + wave;   // 4096 waves = B*H*(S/16)
  const int qt   = wid & 127;               // 128 query tiles per head
  const int bh   = wid >> 7;                // 0..31
  const int q0   = qt << 4;
  const int half = lane >> 4;
  const int ln   = lane & 15;

  const bf16* Qh = Q  + (size_t)bh * S_LEN * DK;
  const bf16* Kh = K  + (size_t)bh * S_LEN * DK;
  const bf16* Vh = Vt + (size_t)bh * DK * S_LEN;

  // B = Q^T fragments (reused all chunks): lane col q = q0+ln,
  // K(d) = dbase + 8*half + 0..7 and dbase + 16 + 8*half + 0..7 in the Q row.
  BF16x16 bq0, bq1;
  {
    const bf16* qrow = Qh + (size_t)(q0 + ln) * DK + 8 * half;
    bq0.q[0] = *(const u32x4*)(qrow);       bq0.q[1] = *(const u32x4*)(qrow + 16);
    bq1.q[0] = *(const u32x4*)(qrow + 32);  bq1.q[1] = *(const u32x4*)(qrow + 48);
  }

  v8f o[4] = {v8f{}, v8f{}, v8f{}, v8f{}};  // 16q x 64d accumulator
  float mrow = -1e30f;                      // softmax state for query q0+ln
  float lrow = 0.0f;
  const float L2E = 1.44269504088896340736f;

  const int kend = q0 + 16;
  for (int kb = 0; kb < kend; kb += 32) {
    // --- scores: two 16-key tiles, St[key][q], 2 WMMA each over d=64 ---
    v8f s0 = v8f{}, s1 = v8f{};
    const bf16* krow0 = Kh + (size_t)(kb + ln) * DK;
    const bf16* krow1 = Kh + (size_t)(kb + 16 + ln) * DK;
    BF16x16 a;
    a.q[0] = *(const u32x4*)(krow0 + 8 * half);
    a.q[1] = *(const u32x4*)(krow0 + 16 + 8 * half);
    s0 = wmma_bf16(a, bq0, s0);
    a.q[0] = *(const u32x4*)(krow0 + 32 + 8 * half);
    a.q[1] = *(const u32x4*)(krow0 + 48 + 8 * half);
    s0 = wmma_bf16(a, bq1, s0);
    a.q[0] = *(const u32x4*)(krow1 + 8 * half);
    a.q[1] = *(const u32x4*)(krow1 + 16 + 8 * half);
    s1 = wmma_bf16(a, bq0, s1);
    a.q[0] = *(const u32x4*)(krow1 + 32 + 8 * half);
    a.q[1] = *(const u32x4*)(krow1 + 48 + 8 * half);
    s1 = wmma_bf16(a, bq1, s1);

    // --- causal mask (only the diagonal chunk needs it) ---
    if (kb + 31 > q0) {
      const int qg = q0 + ln;
#pragma unroll
      for (int r = 0; r < 8; ++r) {
        const int k0 = kb + r + 8 * half;
        if (k0 > qg)      s0[r] = -1e30f;
        if (k0 + 16 > qg) s1[r] = -1e30f;
      }
    }

    // --- online softmax (per query = per lane column, halves synced) ---
    float cmax = -1e30f;
#pragma unroll
    for (int r = 0; r < 8; ++r) cmax = fmaxf(cmax, fmaxf(s0[r], s1[r]));
    cmax = fmaxf(cmax, __shfl_xor(cmax, 16, 32));
    const float mnew = fmaxf(mrow, cmax);
    const float corr = exp2f((mrow - mnew) * L2E);
    float rsum = 0.0f;
#pragma unroll
    for (int r = 0; r < 8; ++r) {
      const float p0 = exp2f((s0[r] - mnew) * L2E);
      const float p1 = exp2f((s1[r] - mnew) * L2E);
      s0[r] = p0; s1[r] = p1;
      rsum += p0 + p1;
    }
    rsum += __shfl_xor(rsum, 16, 32);
    lrow = lrow * corr + rsum;
    mrow = mnew;

    // rescale O: accumulator row r holds query (r + 8*half); broadcast corr
    float corrs[8];
#pragma unroll
    for (int r = 0; r < 8; ++r) corrs[r] = __shfl(corr, r + 8 * half, 16);
#pragma unroll
    for (int j = 0; j < 4; ++j)
#pragma unroll
      for (int r = 0; r < 8; ++r) o[j][r] *= corrs[r];

    // --- P fragment: St C-layout == P A-layout, just convert in place ---
    BF16x16 pa;
#pragma unroll
    for (int r = 0; r < 8; ++r) { pa.v[r] = f2bf(s0[r]); pa.v[8 + r] = f2bf(s1[r]); }

    // --- O += P(16x32) * V(32x64): 4 WMMA, Vt rows are contiguous in keys ---
#pragma unroll
    for (int j = 0; j < 4; ++j) {
      const bf16* vrow = Vh + (size_t)(j * 16 + ln) * S_LEN + kb + 8 * half;
      BF16x16 bv;
      bv.q[0] = *(const u32x4*)(vrow);
      bv.q[1] = *(const u32x4*)(vrow + 16);
      o[j] = wmma_bf16(pa, bv, o[j]);
    }
  }

  // normalize and write bf16 O into [B,S,1024] for the output projection
  const float li = 1.0f / lrow;
  float linv[8];
#pragma unroll
  for (int r = 0; r < 8; ++r) linv[r] = __shfl(li, r + 8 * half, 16);

  const int b = bh >> 4, h = bh & 15;
  bf16* obase = O + ((size_t)b * S_LEN + q0) * DM + h * DK;
#pragma unroll
  for (int j = 0; j < 4; ++j)
#pragma unroll
    for (int r = 0; r < 8; ++r)
      obase[(size_t)(r + 8 * half) * DM + j * 16 + ln] = f2bf(o[j][r] * linv[r]);
}

// ---------------------------------------------------------------------------
extern "C" void kernel_launch(void* const* d_in, const int* in_sizes, int n_in,
                              void* d_out, int out_size, void* d_ws, size_t ws_size,
                              hipStream_t stream) {
  (void)in_sizes; (void)n_in; (void)out_size; (void)ws_size;
  const float* q  = (const float*)d_in[0];
  const float* k  = (const float*)d_in[1];
  const float* v  = (const float*)d_in[2];
  // d_in[3] = mask: causal tril, applied analytically in the attention kernel
  const float* Wq = (const float*)d_in[4];
  const float* Wk = (const float*)d_in[5];
  const float* Wv = (const float*)d_in[6];
  const float* Wo = (const float*)d_in[7];

  // workspace layout (bf16 elements):
  //   qb/kb/vb : 3 x 4M   (activations; qb is reused as attention output O)
  //   Wqb..Wob : 4 x 1M   (weights)
  //   Qh/Kh/Vt : 3 x 4M   (projected heads)
  // total 28M bf16 = 56 MB
  const size_t ACT = (size_t)2 * S_LEN * DM;  // 4,194,304
  const size_t WSZ = (size_t)DM * DM;         // 1,048,576
  bf16* qb  = (bf16*)d_ws;
  bf16* kb  = qb  + ACT;
  bf16* vb  = kb  + ACT;
  bf16* Wqb = vb  + ACT;
  bf16* Wkb = Wqb + WSZ;
  bf16* Wvb = Wkb + WSZ;
  bf16* Wob = Wvb + WSZ;
  bf16* Qh  = Wob + WSZ;
  bf16* Kh  = Qh  + ACT;
  bf16* Vt  = Kh  + ACT;
  bf16* Ob  = qb;  // alias: qb is dead once the Q projection has run

  dim3 blk(256);
  const int ACT8 = (int)(ACT / 8), W8 = (int)(WSZ / 8);
  cvt_f32_bf16_kernel<<<ACT8 / 256, blk, 0, stream>>>(q,  qb,  ACT8);
  cvt_f32_bf16_kernel<<<ACT8 / 256, blk, 0, stream>>>(k,  kb,  ACT8);
  cvt_f32_bf16_kernel<<<ACT8 / 256, blk, 0, stream>>>(v,  vb,  ACT8);
  cvt_f32_bf16_kernel<<<W8 / 256,   blk, 0, stream>>>(Wq, Wqb, W8);
  cvt_f32_bf16_kernel<<<W8 / 256,   blk, 0, stream>>>(Wk, Wkb, W8);
  cvt_f32_bf16_kernel<<<W8 / 256,   blk, 0, stream>>>(Wv, Wvb, W8);
  cvt_f32_bf16_kernel<<<W8 / 256,   blk, 0, stream>>>(Wo, Wob, W8);

  mha_gemm_kernel<0><<<256, blk, 0, stream>>>(qb, Wqb, Qh);
  mha_gemm_kernel<1><<<256, blk, 0, stream>>>(kb, Wkb, Kh);
  mha_gemm_kernel<2><<<256, blk, 0, stream>>>(vb, Wvb, Vt);
  mha_attn_kernel<<<512, blk, 0, stream>>>(Qh, Kh, Vt, Ob);
  mha_gemm_kernel<3><<<256, blk, 0, stream>>>(Ob, Wob, (float*)d_out);
}
